// SuperPoint_47201690583719
// MI455X (gfx1250) — matrix operand order repaired and checked by reading, not compiled, not run on hardware
//
#include <hip/hip_runtime.h>
#include <hip/hip_bf16.h>
#include <math.h>

typedef __attribute__((ext_vector_type(16))) _Float16 v16h;
typedef __attribute__((ext_vector_type(8)))  float    v8f;
typedef _Float16 half_t;

#define BN_EPS 1e-5f
#define KP_THR 0.005f

// ---------------------------------------------------------------------------
// Weight packing: OIHW f32 -> CDNA5 WMMA A-fragment layout, f16, BN-scale folded.
// k index = (ky*3+kx)*Cin + ci  (KHW==1 -> k = ci)
// fragment element i = ((ct*kChunks + kc)*32 + lane)*16 + e
// ---------------------------------------------------------------------------
__global__ void pack_weights(const float* __restrict__ w, const float* __restrict__ cb,
                             const float* __restrict__ g, const float* __restrict__ bb,
                             const float* __restrict__ mean, const float* __restrict__ var,
                             half_t* __restrict__ wpack, float* __restrict__ shift,
                             int Cout, int CoutPad, int Cin, int KHW) {
  const int K = Cin * KHW;
  const int kChunks = K >> 5;
  const int coTiles = CoutPad >> 4;
  const long total = (long)coTiles * kChunks * 32 * 16;
  long stride = (long)gridDim.x * blockDim.x;
  for (long i = blockIdx.x * (long)blockDim.x + threadIdx.x; i < total; i += stride) {
    int e    = (int)(i & 15);
    int lane = (int)((i >> 4) & 31);
    long id  = i >> 9;
    int kc = (int)(id % kChunks);
    int ct = (int)(id / kChunks);
    int r = e >> 1, t = e & 1;
    int M = (lane < 16) ? lane : lane - 16;
    int klo;
    if (lane < 16) klo = 2*r + t;
    else           klo = 16 + 2*r + t;
    int k  = kc*32 + klo;
    int co = ct*16 + M;
    float val = 0.f;
    if (co < Cout) {
      int kpos = k / Cin, ci = k - kpos*Cin;
      float s = g[co] * rsqrtf(var[co] + BN_EPS);
      val = w[((long)co*Cin + ci)*KHW + kpos] * s;
    }
    wpack[i] = (half_t)val;
  }
  for (int co = blockIdx.x * blockDim.x + threadIdx.x; co < CoutPad;
       co += gridDim.x * blockDim.x) {
    float sh = 0.f;
    if (co < Cout) {
      float s = g[co] * rsqrtf(var[co] + BN_EPS);
      sh = cb[co]*s + bb[co] - mean[co]*s;
    }
    shift[co] = sh;
  }
}

// ---------------------------------------------------------------------------
// First conv: Cin=1, 3x3, BN+ReLU fused, f32 NCHW in -> f16 NHWC(64) out
// ---------------------------------------------------------------------------
__global__ __launch_bounds__(64)
void conv_first(const float* __restrict__ x, const float* __restrict__ w,
                const float* __restrict__ cb, const float* __restrict__ g,
                const float* __restrict__ bb, const float* __restrict__ mean,
                const float* __restrict__ var, half_t* __restrict__ out,
                int H, int W) {
  int co = threadIdx.x;             // 0..63
  int p = blockIdx.x;               // pixel within image
  int n = blockIdx.y;
  int xx = p % W, yy = p / W;
  float s = g[co] * rsqrtf(var[co] + BN_EPS);
  float acc = cb[co]*s + bb[co] - mean[co]*s;
  const float* xb = x + (long)n*H*W;
  #pragma unroll
  for (int ky = 0; ky < 3; ++ky) {
    int iy = yy + ky - 1;
    if (iy < 0 || iy >= H) continue;
    #pragma unroll
    for (int kx = 0; kx < 3; ++kx) {
      int ix = xx + kx - 1;
      if (ix < 0 || ix >= W) continue;
      acc += xb[iy*W + ix] * (w[co*9 + ky*3 + kx] * s);
    }
  }
  out[((long)n*H*W + p)*64 + co] = (half_t)fmaxf(acc, 0.f);
}

// ---------------------------------------------------------------------------
// Implicit-GEMM conv via V_WMMA_F32_16X16X32_F16 with 2x2 register tiling.
// Compile-time CIN / KHW / WIDTH -> all divisions become shifts/mul-shift.
// in:  f16 NHWC [N,H,W,CIN]; out: f16 NHWC [N,H,W,Cout], optional ReLU.
// CoutPad multiple of 64. Block: 256 threads = 8 waves = 2 co-rows x 4 px-cols.
// Each wave: 32co x 32px (2 A frags x 2 B frags -> 4 WMMAs / K-chunk).
// ---------------------------------------------------------------------------
template <int CIN, int KHW, int WIDTH>
__global__ __launch_bounds__(256)
void conv_wmma(const half_t* __restrict__ in, const half_t* __restrict__ wpack,
               const float* __restrict__ shift, half_t* __restrict__ out,
               int H, int Cout, int CoutPad, int relu) {
  constexpr int kChunks = (CIN * KHW) >> 5;
  const int lane = threadIdx.x & 31;
  const int wave = threadIdx.x >> 5;
  const int ct0 = blockIdx.z * 4 + (wave >> 2) * 2;   // first of 2 co tiles
  const int HW = H * WIDTH;
  const int p0 = blockIdx.x * 128 + (wave & 3) * 32;

  // two pixel tiles for this wave
  const int pA = p0 + (lane & 15);
  const int pB = pA + 16;
  const bool vA = (pA < HW), vB = (pB < HW);
  const int pcA = vA ? pA : (HW - 1);
  const int pcB = vB ? pB : (HW - 1);
  const int pxA = pcA % WIDTH, pyA = pcA / WIDTH;
  const int pxB = pcB % WIDTH, pyB = pcB / WIDTH;
  const int ciLane = (lane < 16) ? 0 : 16;

  v8f acc[2][2] = {};
  const half_t* w0  = wpack + ((ct0 * kChunks) << 9);
  const half_t* w1  = w0 + (kChunks << 9);
  const half_t* inb = in + (long)blockIdx.y * HW * CIN;

  #pragma unroll 2
  for (int kc = 0; kc < kChunks; ++kc) {
    const int lo = (kc << 9) + (lane << 4);
    v16h a0 = *(const v16h*)(w0 + lo);
    v16h a1 = *(const v16h*)(w1 + lo);

    const int kbase = kc << 5;
    const int kpos  = kbase / CIN;               // shift (CIN power of two)
    const int ci0   = (kbase & (CIN - 1)) + ciLane;
    int dy = 0, dx = 0;
    if (KHW == 9) { dy = kpos / 3 - 1; dx = kpos - (kpos / 3) * 3 - 1; }

    v16h b0 = {}, b1 = {};
    {
      int iy = pyA + dy, ix = pxA + dx;
      bool ok = vA && (iy >= 0) && (iy < H) && (ix >= 0) && (ix < WIDTH);
      if (ok) b0 = *(const v16h*)(inb + (iy * WIDTH + ix) * CIN + ci0);
    }
    {
      int iy = pyB + dy, ix = pxB + dx;
      bool ok = vB && (iy >= 0) && (iy < H) && (ix >= 0) && (ix < WIDTH);
      if (ok) b1 = *(const v16h*)(inb + (iy * WIDTH + ix) * CIN + ci0);
    }

    acc[0][0] = __builtin_amdgcn_wmma_f32_16x16x32_f16(false, a0, false, b0,
                                                       (short)0, acc[0][0], false, false);
    acc[0][1] = __builtin_amdgcn_wmma_f32_16x16x32_f16(false, a0, false, b1,
                                                       (short)0, acc[0][1], false, false);
    acc[1][0] = __builtin_amdgcn_wmma_f32_16x16x32_f16(false, a1, false, b0,
                                                       (short)0, acc[1][0], false, false);
    acc[1][1] = __builtin_amdgcn_wmma_f32_16x16x32_f16(false, a1, false, b1,
                                                       (short)0, acc[1][1], false, false);
  }

  #pragma unroll
  for (int ti = 0; ti < 2; ++ti) {
    const int coBase = (ct0 + ti) * 16 + ((lane < 16) ? 0 : 8);
    #pragma unroll
    for (int tj = 0; tj < 2; ++tj) {
      int p = p0 + tj * 16 + (lane & 15);
      if (p >= HW) continue;
      long base = ((long)blockIdx.y * HW + p) * Cout + coBase;
      if (((Cout & 7) == 0) && (coBase + 8 <= Cout)) {
        union { float4 f4; half_t hh[8]; } u;
        #pragma unroll
        for (int r = 0; r < 8; ++r) {
          float o = acc[ti][tj][r] + shift[coBase + r];
          if (relu) o = fmaxf(o, 0.f);
          u.hh[r] = (half_t)o;
        }
        *(float4*)(out + base) = u.f4;
      } else {
        #pragma unroll
        for (int r = 0; r < 8; ++r) {
          int co = coBase + r;
          if (co < Cout) {
            float o = acc[ti][tj][r] + shift[co];
            if (relu) o = fmaxf(o, 0.f);
            out[base + r] = (half_t)o;
          }
        }
      }
    }
  }
}

// ---------------------------------------------------------------------------
// 2x2 max-pool, f16 NHWC
// ---------------------------------------------------------------------------
__global__ void maxpool2_nhwc(const half_t* __restrict__ in, half_t* __restrict__ out,
                              int N, int H, int W, int C) {
  const int Ho = H >> 1, Wo = W >> 1;
  long total = (long)N * Ho * Wo * C;
  long stride = (long)gridDim.x * blockDim.x;
  for (long i = blockIdx.x * (long)blockDim.x + threadIdx.x; i < total; i += stride) {
    int c = (int)(i % C); long t = i / C;
    int x = (int)(t % Wo); t /= Wo;
    int y = (int)(t % Ho); int n = (int)(t / Ho);
    const half_t* pp = in + (((long)n*H + 2*y)*W + 2*x)*C + c;
    float a = (float)pp[0], b = (float)pp[C];
    float d = (float)pp[(long)W*C], e = (float)pp[(long)W*C + C];
    out[i] = (half_t)fmaxf(fmaxf(a, b), fmaxf(d, e));
  }
}

// ---------------------------------------------------------------------------
// Softmax over 65 channels; drop dustbin; depth-to-space 8x8 -> f32 [N,480,640]
// ---------------------------------------------------------------------------
__global__ void softmax_scores(const half_t* __restrict__ semi, float* __restrict__ scores,
                               int N, int h, int w) {
  int idx = blockIdx.x * blockDim.x + threadIdx.x;
  int total = N * h * w;
  if (idx >= total) return;
  int x = idx % w; int t = idx / w; int y = t % h; int n = t / h;
  const half_t* s = semi + (long)idx * 65;
  float mx = -1e30f;
  for (int i = 0; i < 65; ++i) mx = fmaxf(mx, (float)s[i]);
  float e[64], sum = 0.f;
  for (int i = 0; i < 65; ++i) {
    float ev = expf((float)s[i] - mx);
    if (i < 64) e[i] = ev;
    sum += ev;
  }
  float inv = 1.f / sum;
  const int H = h * 8, W = w * 8;
  for (int r = 0; r < 8; ++r)
    for (int c = 0; c < 8; ++c)
      scores[((long)n*H + y*8 + r)*W + x*8 + c] = e[r*8 + c] * inv;
}

// ---------------------------------------------------------------------------
// Separable 9x9 max pooling (radius 4), f32 [N*H, W]
// ---------------------------------------------------------------------------
__global__ void k_rowmax(const float* __restrict__ in, float* __restrict__ out,
                         int W, long total) {
  long stride = (long)gridDim.x * blockDim.x;
  for (long t = blockIdx.x * (long)blockDim.x + threadIdx.x; t < total; t += stride) {
    int x = (int)(t % W); long r = t / W;
    const float* row = in + r * W;
    int lo = max(x - 4, 0), hi = min(x + 4, W - 1);
    float m = -1e30f;
    for (int i = lo; i <= hi; ++i) m = fmaxf(m, row[i]);
    out[t] = m;
  }
}
__global__ void k_colmax(const float* __restrict__ in, float* __restrict__ out,
                         int H, int W, long total) {
  long stride = (long)gridDim.x * blockDim.x;
  for (long t = blockIdx.x * (long)blockDim.x + threadIdx.x; t < total; t += stride) {
    int x = (int)(t % W); long r = t / W;
    int y = (int)(r % H); long n = r / H;
    int lo = max(y - 4, 0), hi = min(y + 4, H - 1);
    const float* base = in + (n*H)*W + x;
    float m = -1e30f;
    for (int i = lo; i <= hi; ++i) m = fmaxf(m, base[(long)i*W]);
    out[t] = m;
  }
}
__global__ void k_eqmask(const float* __restrict__ s, const float* __restrict__ P,
                         float* __restrict__ m, long total) {
  long stride = (long)gridDim.x * blockDim.x;
  for (long t = blockIdx.x * (long)blockDim.x + threadIdx.x; t < total; t += stride)
    m[t] = (s[t] == P[t]) ? 1.f : 0.f;
}
__global__ void k_supp(const float* __restrict__ s, const float* __restrict__ Pm,
                       float* __restrict__ supp, float* __restrict__ ss, long total) {
  long stride = (long)gridDim.x * blockDim.x;
  for (long t = blockIdx.x * (long)blockDim.x + threadIdx.x; t < total; t += stride) {
    bool sp = Pm[t] > 0.f;
    supp[t] = sp ? 1.f : 0.f;
    ss[t]   = sp ? 0.f : s[t];
  }
}
__global__ void k_update(float* __restrict__ m, const float* __restrict__ ss,
                         const float* __restrict__ Pss, const float* __restrict__ supp,
                         long total) {
  long stride = (long)gridDim.x * blockDim.x;
  for (long t = blockIdx.x * (long)blockDim.x + threadIdx.x; t < total; t += stride) {
    if (m[t] == 0.f)
      m[t] = ((ss[t] == Pss[t]) && (supp[t] == 0.f)) ? 1.f : 0.f;
  }
}

// ---------------------------------------------------------------------------
// Top-K: monotonic float key + bitonic sort
// ---------------------------------------------------------------------------
__device__ __forceinline__ unsigned fmap(float f) {
  unsigned u = __float_as_uint(f);
  return (u & 0x80000000u) ? ~u : (u | 0x80000000u);
}
__device__ __forceinline__ float funmap(unsigned m) {
  unsigned u = (m & 0x80000000u) ? (m & 0x7FFFFFFFu) : ~m;
  return __uint_as_float(u);
}

#define SORT_N 65536

__global__ void k_init_keys(unsigned long long* __restrict__ keys, int* __restrict__ counts,
                            int N) {
  long total = (long)N * SORT_N;
  long stride = (long)gridDim.x * blockDim.x;
  const unsigned long long padHi = ((unsigned long long)fmap(-1.0f)) << 32;
  for (long t = blockIdx.x * (long)blockDim.x + threadIdx.x; t < total; t += stride) {
    unsigned slot = (unsigned)(t % SORT_N);
    keys[t] = padHi | (unsigned long long)(0xFFFFFFFFu - slot);
    if (t < N) counts[t] = 0;
  }
}

__global__ void k_collect(const float* __restrict__ s, const float* __restrict__ mask,
                          unsigned long long* __restrict__ keys, int* __restrict__ counts,
                          int N, int H, int W) {
  long total = (long)N * H * W;
  long stride = (long)gridDim.x * blockDim.x;
  for (long t = blockIdx.x * (long)blockDim.x + threadIdx.x; t < total; t += stride) {
    int x = (int)(t % W); long r = t / W;
    int y = (int)(r % H); int n = (int)(r / H);
    float sv = s[t];
    bool valid = (mask[t] != 0.f) && (sv > KP_THR) &&
                 (x >= 4) && (x < W - 4) && (y >= 4) && (y < H - 4);
    if (valid) {
      int pos = atomicAdd(&counts[n], 1);
      if (pos < SORT_N) {
        unsigned fi = (unsigned)(y * W + x);
        keys[(long)n*SORT_N + pos] =
            (((unsigned long long)fmap(sv)) << 32) | (unsigned long long)(0xFFFFFFFFu - fi);
      }
    }
  }
}

__global__ void k_bitonic(unsigned long long* __restrict__ keys, int N, int k, int j) {
  long tid = blockIdx.x * (long)blockDim.x + threadIdx.x;
  long total = (long)N * SORT_N;
  if (tid >= total) return;
  int b = (int)(tid / SORT_N), i = (int)(tid % SORT_N);
  int l = i ^ j;
  if (l > i) {
    unsigned long long* Kp = keys + (long)b * SORT_N;
    unsigned long long a = Kp[i], c = Kp[l];
    bool descDir = ((i & k) == 0);
    if (descDir ? (a < c) : (a > c)) { Kp[i] = c; Kp[l] = a; }
  }
}

// kpts at [0,8192), scores at [8192,12288), descs at [12288,1060864), mask at [1060864,...)
#define OUT_SCORES 8192
#define OUT_DESC   12288
#define OUT_MASK   1060864

__global__ void k_write_outputs(const unsigned long long* __restrict__ keys,
                                float* __restrict__ out, int N, int W) {
  int t = blockIdx.x * blockDim.x + threadIdx.x;
  if (t >= N * 1024) return;
  int n = t >> 10, i = t & 1023;
  unsigned long long key = keys[(long)n*SORT_N + i];
  float sc = funmap((unsigned)(key >> 32));
  unsigned idx = 0xFFFFFFFFu - (unsigned)(key & 0xFFFFFFFFu);
  out[((long)n*1024 + i)*2 + 0] = (float)(idx % (unsigned)W);
  out[((long)n*1024 + i)*2 + 1] = (float)(idx / (unsigned)W);
  out[OUT_SCORES + t] = sc;
  out[OUT_MASK + t]   = (sc > KP_THR) ? 1.f : 0.f;
}

// ---------------------------------------------------------------------------
// Per-pixel L2 normalization of descriptor map, f16 NHWC [N,h,w,256], in place
// ---------------------------------------------------------------------------
__global__ __launch_bounds__(256)
void desc_norm(half_t* __restrict__ d) {
  long p = blockIdx.x;
  int c = threadIdx.x;
  __shared__ float red[256];
  float v = (float)d[p*256 + c];
  red[c] = v * v;
  __syncthreads();
  for (int s = 128; s > 0; s >>= 1) { if (c < s) red[c] += red[c + s]; __syncthreads(); }
  float inv = rsqrtf(fmaxf(red[0], 1e-24f));
  d[p*256 + c] = (half_t)(v * inv);
}

__device__ __forceinline__ float dtap(const half_t* dmap, int n, int yy, int xx,
                                      int c, int h, int w) {
  bool ok = (yy >= 0) && (yy < h) && (xx >= 0) && (xx < w);
  int yc = min(max(yy, 0), h - 1), xc = min(max(xx, 0), w - 1);
  float v = (float)dmap[(((long)n*h + yc)*w + xc)*256 + c];
  return ok ? v : 0.f;
}

// ---------------------------------------------------------------------------
// Bilinear descriptor sampling + renormalize. grid=(1024, N), block=256
// ---------------------------------------------------------------------------
__global__ __launch_bounds__(256)
void sample_desc(const unsigned long long* __restrict__ keys, const half_t* __restrict__ dmap,
                 float* __restrict__ out, int h, int w) {
  int i = blockIdx.x, n = blockIdx.y, c = threadIdx.x;
  unsigned long long key = keys[(long)n*SORT_N + i];
  unsigned idx = 0xFFFFFFFFu - (unsigned)(key & 0xFFFFFFFFu);
  const int W = w * 8, H = h * 8;
  float kx = (float)(idx % (unsigned)W);
  float ky = (float)(idx / (unsigned)W);
  float px = (kx - 3.5f) * ((float)(w - 1) / ((float)W - 4.5f));
  float py = (ky - 3.5f) * ((float)(h - 1) / ((float)H - 4.5f));
  float x0f = floorf(px), y0f = floorf(py);
  float wx1 = px - x0f, wy1 = py - y0f;
  float wx0 = 1.f - wx1, wy0 = 1.f - wy1;
  int x0 = (int)x0f, y0 = (int)y0f;
  float acc = dtap(dmap, n, y0,     x0,     c, h, w) * (wy0 * wx0)
            + dtap(dmap, n, y0,     x0 + 1, c, h, w) * (wy0 * wx1)
            + dtap(dmap, n, y0 + 1, x0,     c, h, w) * (wy1 * wx0)
            + dtap(dmap, n, y0 + 1, x0 + 1, c, h, w) * (wy1 * wx1);
  __shared__ float red[256];
  red[c] = acc * acc;
  __syncthreads();
  for (int s = 128; s > 0; s >>= 1) { if (c < s) red[c] += red[c + s]; __syncthreads(); }
  float inv = 1.f / fmaxf(sqrtf(red[0]), 1e-12f);
  out[OUT_DESC + ((long)(n*256 + c))*1024 + i] = acc * inv;
}

// ---------------------------------------------------------------------------
// Host orchestration
// ---------------------------------------------------------------------------
extern "C" void kernel_launch(void* const* d_in, const int* in_sizes, int n_in,
                              void* d_out, int out_size, void* d_ws, size_t ws_size,
                              hipStream_t stream) {
  (void)in_sizes; (void)n_in; (void)out_size; (void)ws_size;
  const int N = 4, H = 480, W = 640, h = 60, w = 80;
  const long HW = (long)H * W;

  // Input flattening: setup_inputs() insertion order.
  // 0: x
  // per double_conv (12): w1,b1,g,b,m,v, w2,b2,g,b,m,v
  // 1..12 inc, 13..24 down1, 25..36 down2, 37..48 down3
  // 49 wPa, 50 bPa, 51-54 bnPa, 55 wPb, 56 bPb, 57-60 bnPb,
  // 61 wDa, 62 bDa, 63-66 bnDa, 67 wDb, 68 bDb, 69-72 bnDb
  auto F = [&](int i) { return (const float*)d_in[i]; };

  struct Layer { int wi; int Cin, Cout, CoutPad, KHW; size_t wOff, sOff; };
  Layer L[11] = {
    { 7,  64,  64,  64, 9, 0, 0},  // inc conv2
    {13,  64,  64,  64, 9, 0, 0},  // down1 conv1
    {19,  64,  64,  64, 9, 0, 0},  // down1 conv2
    {25,  64, 128, 128, 9, 0, 0},  // down2 conv1
    {31, 128, 128, 128, 9, 0, 0},  // down2 conv2
    {37, 128, 128, 128, 9, 0, 0},  // down3 conv1
    {43, 128, 128, 128, 9, 0, 0},  // down3 conv2
    {49, 128, 256, 256, 9, 0, 0},  // wPa
    {55, 256,  65, 128, 1, 0, 0},  // wPb (padded to 128 for 64-co blocks)
    {61, 128, 256, 256, 9, 0, 0},  // wDa
    {67, 256, 256, 256, 1, 0, 0},  // wDb
  };

  // workspace carving
  size_t off = 0;
  auto carve = [&](size_t bytes) {
    size_t o = off; off += (bytes + 255) & ~(size_t)255; return o;
  };
  char* ws = (char*)d_ws;
  size_t bufA_o  = carve((size_t)N*HW*64*2);           // 157 MB f16 NHWC max
  size_t bufB_o  = carve((size_t)N*HW*64*2);
  size_t semi_o  = carve((size_t)N*h*w*65*2);
  size_t descm_o = carve((size_t)N*h*w*256*2);
  size_t sS_o    = carve((size_t)N*HW*4);
  size_t tT_o    = carve((size_t)N*HW*4);
  size_t tP_o    = carve((size_t)N*HW*4);
  size_t tM_o    = carve((size_t)N*HW*4);
  size_t tSS_o   = carve((size_t)N*HW*4);
  size_t tSupp_o = carve((size_t)N*HW*4);
  size_t keys_o  = carve((size_t)N*SORT_N*8);
  size_t cnt_o   = carve(256);
  for (int i = 0; i < 11; ++i) {
    size_t K = (size_t)L[i].Cin * L[i].KHW;
    L[i].wOff = carve((size_t)L[i].CoutPad * K * 2);
    L[i].sOff = carve((size_t)L[i].CoutPad * 4);
  }

  half_t* bufA  = (half_t*)(ws + bufA_o);
  half_t* bufB  = (half_t*)(ws + bufB_o);
  half_t* semiB = (half_t*)(ws + semi_o);
  half_t* descB = (half_t*)(ws + descm_o);
  float* sS = (float*)(ws + sS_o);   float* tT    = (float*)(ws + tT_o);
  float* tP = (float*)(ws + tP_o);   float* tM    = (float*)(ws + tM_o);
  float* tSS = (float*)(ws + tSS_o); float* tSupp = (float*)(ws + tSupp_o);
  unsigned long long* keys = (unsigned long long*)(ws + keys_o);
  int* counts = (int*)(ws + cnt_o);

  // 1) pack all WMMA-layer weights
  for (int i = 0; i < 11; ++i) {
    int wi = L[i].wi;
    pack_weights<<<256, 256, 0, stream>>>(
        F(wi), F(wi+1), F(wi+2), F(wi+3), F(wi+4), F(wi+5),
        (half_t*)(ws + L[i].wOff), (float*)(ws + L[i].sOff),
        L[i].Cout, L[i].CoutPad, L[i].Cin, L[i].KHW);
  }

#define CONV(li, CIN_, KHW_, WW_, inp, outp, HH_, relu_)                         \
  do {                                                                           \
    Layer& l = L[li];                                                            \
    dim3 grid((unsigned)(((HH_)*(WW_) + 127) / 128), N,                          \
              (unsigned)(l.CoutPad / 64));                                       \
    conv_wmma<CIN_, KHW_, WW_><<<grid, 256, 0, stream>>>(                        \
        inp, (const half_t*)(ws + l.wOff), (const float*)(ws + l.sOff), outp,    \
        HH_, l.Cout, l.CoutPad, relu_);                                          \
  } while (0)

  auto pool = [&](const half_t* in, half_t* outp, int Hh, int Ww, int C) {
    maxpool2_nhwc<<<8192, 256, 0, stream>>>(in, outp, N, Hh, Ww, C);
  };

  // 2) backbone
  conv_first<<<dim3((unsigned)HW, N), 64, 0, stream>>>(
      F(0), F(1), F(2), F(3), F(4), F(5), F(6), bufA, H, W);
  CONV(0, 64, 9, 640, bufA, bufB, 480, 1);
  pool(bufB, bufA, 480, 640, 64);
  CONV(1, 64, 9, 320, bufA, bufB, 240, 1);
  CONV(2, 64, 9, 320, bufB, bufA, 240, 1);
  pool(bufA, bufB, 240, 320, 64);
  CONV(3, 64, 9, 160, bufB, bufA, 120, 1);
  CONV(4, 128, 9, 160, bufA, bufB, 120, 1);
  pool(bufB, bufA, 120, 160, 128);
  CONV(5, 128, 9, 80, bufA, bufB, 60, 1);
  CONV(6, 128, 9, 80, bufB, bufA, 60, 1);   // x4 in bufA
  // 3) heads
  CONV(7, 128, 9, 80, bufA, bufB, 60, 1);   // cPa
  CONV(8, 256, 1, 80, bufB, semiB, 60, 0);  // semi (65ch, BN only)
  CONV(9, 128, 9, 80, bufA, bufB, 60, 1);   // cDa
  CONV(10, 256, 1, 80, bufB, descB, 60, 0); // desc map (BN only)
#undef CONV
  desc_norm<<<(unsigned)(N*h*w), 256, 0, stream>>>(descB);

  // 4) softmax -> full-res score map
  softmax_scores<<<(N*h*w + 255) / 256, 256, 0, stream>>>(semiB, sS, N, h, w);

  // 5) simple_nms (separable 9x9 max pools)
  long total = (long)N * HW;
  const int GB = 4800, TB = 256;
  k_rowmax<<<GB, TB, 0, stream>>>(sS, tT, W, total);
  k_colmax<<<GB, TB, 0, stream>>>(tT, tP, H, W, total);
  k_eqmask<<<GB, TB, 0, stream>>>(sS, tP, tM, total);
  for (int it = 0; it < 2; ++it) {
    k_rowmax<<<GB, TB, 0, stream>>>(tM, tT, W, total);
    k_colmax<<<GB, TB, 0, stream>>>(tT, tP, H, W, total);
    k_supp<<<GB, TB, 0, stream>>>(sS, tP, tSupp, tSS, total);
    k_rowmax<<<GB, TB, 0, stream>>>(tSS, tT, W, total);
    k_colmax<<<GB, TB, 0, stream>>>(tT, tP, H, W, total);
    k_update<<<GB, TB, 0, stream>>>(tM, tSS, tP, tSupp, total);
  }

  // 6) candidate collection + bitonic top-k
  k_init_keys<<<1024, 256, 0, stream>>>(keys, counts, N);
  k_collect<<<GB, TB, 0, stream>>>(sS, tM, keys, counts, N, H, W);
  for (int k = 2; k <= SORT_N; k <<= 1)
    for (int j = k >> 1; j > 0; j >>= 1)
      k_bitonic<<<(N * SORT_N) / 256, 256, 0, stream>>>(keys, N, k, j);

  // 7) outputs
  float* outF = (float*)d_out;
  k_write_outputs<<<(N * 1024 + 255) / 256, 256, 0, stream>>>(keys, outF, N, W);
  sample_desc<<<dim3(1024, N), 256, 0, stream>>>(keys, descB, outF, h, w);
}
